// FinancialNoisyTopKGate_15109694948211
// MI455X (gfx1250) — matrix operand order repaired and checked by reading, not compiled, not run on hardware
//
#include <hip/hip_runtime.h>
#include <hip/hip_bf16.h>
#include <cstdint>

typedef __attribute__((ext_vector_type(16))) __bf16 v16bf;
typedef __attribute__((ext_vector_type(8)))  float  v8f;
typedef unsigned int u32t;

#define BROWS 16384
#define NEXP  64

// ---------- helpers ----------
__device__ __forceinline__ unsigned short f2bf_bits(float f) {
    u32t u = __float_as_uint(f);
    u += 0x7FFFu + ((u >> 16) & 1u);   // round-to-nearest-even
    return (unsigned short)(u >> 16);
}

// A-fragment LDS pair index, contiguous k order used by this code (B loads match):
// lanes 0-15 : V0..7 -> K=0..15
// lanes16-31 : V0..7 -> K=16..31
__device__ __forceinline__ int a_pair_idx(int vv, int half) {
    return vv + 8 * half;
}

// gfx1250 async global->LDS copy (ASYNCcnt-tracked), per-lane 8 bytes
__device__ __forceinline__ void async_ld_b64(unsigned int lds_off, const void* gaddr) {
    asm volatile("global_load_async_to_lds_b64 %0, %1, off"
                 :: "v"(lds_off), "v"(gaddr) : "memory");
}
__device__ __forceinline__ void wait_async0() {
    asm volatile("s_wait_asynccnt 0x0" ::: "memory");
}
// low 32 bits of generic address of a __shared__ object == LDS byte offset
__device__ __forceinline__ unsigned int lds_off32(const void* p) {
    return (unsigned int)(uintptr_t)p;
}

// ---------- weight convert + pad (fp32 -> bf16) ----------
__global__ void cvt_pad_kernel(const float* __restrict__ src, unsigned short* __restrict__ dst,
                               int R, int C, int RP, int CP) {
    int total = RP * CP;
    for (int i = blockIdx.x * blockDim.x + threadIdx.x; i < total; i += gridDim.x * blockDim.x) {
        int r = i / CP, c = i % CP;
        float v = (r < R && c < C) ? src[r * C + c] : 0.f;
        dst[i] = f2bf_bits(v);
    }
}

// pack the 4 branch LN params into contiguous [64] arrays
__global__ void init_kernel(const float* vb, const float* vg, const float* vbb,
                            const float* tb, const float* tg, const float* tbb,
                            const float* mb, const float* mg, const float* mbb,
                            const float* rb, const float* rg, const float* rbb,
                            float* enc_b, float* enc_g, float* enc_bb) {
    int t = threadIdx.x;            // 64 threads
    int br = t >> 4, i = t & 15;
    const float* Bp[4]  = {vb, tb, mb, rb};
    const float* Gp[4]  = {vg, tg, mg, rg};
    const float* BBp[4] = {vbb, tbb, mbb, rbb};
    enc_b[t]  = Bp[br][i];
    enc_g[t]  = Gp[br][i];
    enc_bb[t] = BBp[br][i];
}

// ---------- branch encoders + router; builds packed bf16 gi[B,608] ----------
__global__ __launch_bounds__(128)
void encoder_kernel(const float* __restrict__ x,                 // [B,512]
                    const unsigned short* __restrict__ Wb,       // [64,512] bf16
                    const float* __restrict__ enc_b, const float* __restrict__ enc_g,
                    const float* __restrict__ enc_bb,
                    const float* __restrict__ rc1_w, const float* __restrict__ rc1_b,
                    const float* __restrict__ rc2_w, const float* __restrict__ rc2_b,
                    const float* __restrict__ ns1_w, const float* __restrict__ ns1_b,
                    const float* __restrict__ ns2_w, const float* __restrict__ ns2_b,
                    unsigned short* __restrict__ gi,             // [B,608] bf16
                    float* __restrict__ rp_out,                  // [B,4]
                    float* __restrict__ nscale_out)              // [B]
{
    __shared__ u32t  ldsA[256];          // 16 rows x 16 bf16-pairs
    __shared__ float mfbuf[16][64];
    __shared__ float hrc[16][32];
    __shared__ float rpraw[16][4];

    const int tid  = threadIdx.x;
    const int wave = tid >> 5;
    const int lane = tid & 31;
    const int n16  = lane & 15;
    const int half = lane >> 4;
    const int row0 = blockIdx.x * 16;

    v8f acc = {0.f,0.f,0.f,0.f,0.f,0.f,0.f,0.f};

    for (int k0 = 0; k0 < 512; k0 += 32) {
        #pragma unroll
        for (int j = 0; j < 2; ++j) {
            int lin = tid * 2 + j;       // 0..255
            int r = lin >> 4, cp = lin & 15;
            const float* xp = x + (size_t)(row0 + r) * 512 + k0 + 2 * cp;
            u32t u = (u32t)f2bf_bits(xp[0]) | ((u32t)f2bf_bits(xp[1]) << 16);
            ldsA[lin] = u;
            *(u32t*)(gi + (size_t)(row0 + r) * 608 + k0 + 2 * cp) = u;  // gi[:,0:512] = bf16(x)
        }
        __syncthreads();
        union { v16bf v; u32t u[8]; } af;
        #pragma unroll
        for (int vv = 0; vv < 8; ++vv) af.u[vv] = ldsA[n16 * 16 + a_pair_idx(vv, half)];
        const unsigned short* wp = Wb + (size_t)(wave * 16 + n16) * 512 + k0 + half * 16;
        v16bf bfv = *(const v16bf*)wp;
        acc = __builtin_amdgcn_wmma_f32_16x16x32_bf16(false, af.v, false, bfv,
                                                      (short)0, acc, false, false);
        __syncthreads();
    }

    // ReLU + per-branch (16-wide) LayerNorm; wave == branch
    int c = wave * 16 + n16;
    float bia = enc_b[c], gg = enc_g[c], bb = enc_bb[c];
    float tv[8], s[8], sq[8];
    #pragma unroll
    for (int r = 0; r < 8; ++r) {
        float v = fmaxf(acc[r] + bia, 0.f);
        tv[r] = v; s[r] = v; sq[r] = v * v;
    }
    #pragma unroll
    for (int m = 1; m < 16; m <<= 1) {
        #pragma unroll
        for (int r = 0; r < 8; ++r) {
            s[r]  += __shfl_xor(s[r],  m, 32);
            sq[r] += __shfl_xor(sq[r], m, 32);
        }
    }
    #pragma unroll
    for (int r = 0; r < 8; ++r) {
        int row = half * 8 + r;
        float mu   = s[r] * (1.f / 16.f);
        float var  = sq[r] * (1.f / 16.f) - mu * mu;
        float rstd = rsqrtf(var + 1e-5f);
        float y = (tv[r] - mu) * rstd * gg + bb;
        mfbuf[row][c] = y;
        gi[(size_t)(row0 + row) * 608 + 512 + c] = f2bf_bits(y);
    }
    __syncthreads();

    // router: rc1 (64->32, relu)
    #pragma unroll
    for (int j = 0; j < 4; ++j) {
        int task = tid * 4 + j;          // 0..511
        int r = task >> 5, o = task & 31;
        float a = rc1_b[o];
        #pragma unroll 8
        for (int i = 0; i < 64; ++i) a += mfbuf[r][i] * rc1_w[o * 64 + i];
        hrc[r][o] = fmaxf(a, 0.f);
    }
    __syncthreads();
    // rc2 (32->4)
    if (tid < 64) {
        int r = tid >> 2, o = tid & 3;
        float a = rc2_b[o];
        #pragma unroll 8
        for (int i = 0; i < 32; ++i) a += hrc[r][i] * rc2_w[o * 32 + i];
        rpraw[r][o] = a;
    }
    __syncthreads();
    // softmax -> rp; pack gi tail; noise scale
    if (tid < 16) {
        int r = tid;
        float m = fmaxf(fmaxf(rpraw[r][0], rpraw[r][1]), fmaxf(rpraw[r][2], rpraw[r][3]));
        float e[4], se = 0.f;
        #pragma unroll
        for (int j = 0; j < 4; ++j) { e[j] = __expf(rpraw[r][j] - m); se += e[j]; }
        float inv = 1.f / se;
        float p[4];
        #pragma unroll
        for (int j = 0; j < 4; ++j) {
            p[j] = e[j] * inv;
            rp_out[(size_t)(row0 + r) * 4 + j] = p[j];
            gi[(size_t)(row0 + r) * 608 + 576 + j] = f2bf_bits(p[j]);
        }
        for (int cc = 580; cc < 608; ++cc) gi[(size_t)(row0 + r) * 608 + cc] = 0;  // K-pad
        float z = ns2_b[0];
        #pragma unroll
        for (int o = 0; o < 16; ++o) {
            float h = ns1_b[o];
            #pragma unroll
            for (int j = 0; j < 4; ++j) h += p[j] * ns1_w[o * 4 + j];
            z += fmaxf(h, 0.f) * ns2_w[o];
        }
        nscale_out[row0 + r] = 1.f / (1.f + __expf(-z));
    }
}

// ---------- WMMA GEMM + bias + LayerNorm(row) + exact GELU, bf16 out ----------
// 32 rows per workgroup (MT=2 row tiles), async double-buffered A staging.
template<int NPAD, int KPAD, int NVALID, int NWAVES, int TPW, int MT>
__global__ __launch_bounds__(NWAVES * 32)
void gemm_ln_gelu_kernel(const unsigned short* __restrict__ A,   // [B,KPAD] bf16
                         const unsigned short* __restrict__ W,   // [NPAD,KPAD] bf16
                         const float* __restrict__ bias,
                         const float* __restrict__ gamma,
                         const float* __restrict__ beta,
                         unsigned short* __restrict__ out)       // [B,NPAD] bf16
{
    constexpr int NTILES = NPAD / 16;
    constexpr int ROWS   = MT * 16;
    static_assert(MT * 128 == NWAVES * 32, "one b64 async chunk per thread");
    __shared__ u32t  ldsA[2][ROWS * 16];          // double-buffered bf16 A tile
    __shared__ float psum[NWAVES][ROWS];
    __shared__ float psq [NWAVES][ROWS];
    __shared__ float smu[ROWS], srstd[ROWS];

    const int tid   = threadIdx.x;
    const int wave  = tid >> 5;
    const int lane  = tid & 31;
    const int n16   = lane & 15;
    const int half  = lane >> 4;
    const int row0  = blockIdx.x * ROWS;
    const int tbase = wave * TPW;
    const unsigned ldsbase = lds_off32(&ldsA[0][0]);

    v8f acc[MT][TPW];
    const v8f vz = {0.f,0.f,0.f,0.f,0.f,0.f,0.f,0.f};
    #pragma unroll
    for (int mt = 0; mt < MT; ++mt)
        #pragma unroll
        for (int t = 0; t < TPW; ++t) acc[mt][t] = vz;

    // each thread owns one 8-byte chunk of the ROWSx32 bf16 tile
    const int ar  = tid >> 3;        // row 0..ROWS-1
    const int ac4 = tid & 7;         // 4-element chunk 0..7
    auto issue = [&](int k0, int buf) {
        const void* g = (const void*)(A + (size_t)(row0 + ar) * KPAD + k0 + 4 * ac4);
        async_ld_b64(ldsbase + (unsigned)(buf * (ROWS * 64) + tid * 8), g);
    };

    issue(0, 0);
    int cur = 0;
    for (int k0 = 0; k0 < KPAD; k0 += 32) {
        wait_async0();
        __syncthreads();                       // buf[cur] visible to all waves
        if (k0 + 32 < KPAD) issue(k0 + 32, cur ^ 1);

        union AF { v16bf v; u32t u[8]; } af[MT];
        #pragma unroll
        for (int mt = 0; mt < MT; ++mt)
            #pragma unroll
            for (int vv = 0; vv < 8; ++vv)
                af[mt].u[vv] = ldsA[cur][(mt * 16 + n16) * 16 + a_pair_idx(vv, half)];

        #pragma unroll
        for (int t = 0; t < TPW; ++t) {
            int tile = tbase + t;
            if (tile < NTILES) {
                const unsigned short* wp = W + (size_t)(tile * 16 + n16) * KPAD + k0 + half * 16;
                v16bf bfv = *(const v16bf*)wp;
                if (k0 + 32 < KPAD) __builtin_prefetch(wp + 32, 0, 1);
                #pragma unroll
                for (int mt = 0; mt < MT; ++mt)
                    acc[mt][t] = __builtin_amdgcn_wmma_f32_16x16x32_bf16(
                        false, af[mt].v, false, bfv, (short)0, acc[mt][t], false, false);
            }
        }
        cur ^= 1;
    }
    __syncthreads();

    // bias + per-row partial sums (C layout: VGPR r -> row r + 8*half, lane%16 -> col)
    float s[MT][8], sq[MT][8];
    #pragma unroll
    for (int mt = 0; mt < MT; ++mt)
        #pragma unroll
        for (int r = 0; r < 8; ++r) { s[mt][r] = 0.f; sq[mt][r] = 0.f; }
    #pragma unroll
    for (int t = 0; t < TPW; ++t) {
        int tile = tbase + t;
        if (tile >= NTILES) continue;
        int c = tile * 16 + n16;
        float bia = (c < NVALID) ? bias[c] : 0.f;
        #pragma unroll
        for (int mt = 0; mt < MT; ++mt)
            #pragma unroll
            for (int r = 0; r < 8; ++r) {
                float v = acc[mt][t][r] + bia;
                acc[mt][t][r] = v;
                if (c < NVALID) { s[mt][r] += v; sq[mt][r] += v * v; }
            }
    }
    #pragma unroll
    for (int m = 1; m < 16; m <<= 1)
        #pragma unroll
        for (int mt = 0; mt < MT; ++mt)
            #pragma unroll
            for (int r = 0; r < 8; ++r) {
                s[mt][r]  += __shfl_xor(s[mt][r],  m, 32);
                sq[mt][r] += __shfl_xor(sq[mt][r], m, 32);
            }
    if (n16 == 0) {
        #pragma unroll
        for (int mt = 0; mt < MT; ++mt)
            #pragma unroll
            for (int r = 0; r < 8; ++r) {
                psum[wave][mt * 16 + half * 8 + r] = s[mt][r];
                psq [wave][mt * 16 + half * 8 + r] = sq[mt][r];
            }
    }
    __syncthreads();
    if (tid < ROWS) {
        float S = 0.f, Q = 0.f;
        #pragma unroll
        for (int w = 0; w < NWAVES; ++w) { S += psum[w][tid]; Q += psq[w][tid]; }
        float mu  = S / (float)NVALID;
        float var = Q / (float)NVALID - mu * mu;
        smu[tid]   = mu;
        srstd[tid] = rsqrtf(var + 1e-5f);
    }
    __syncthreads();
    // LN + exact GELU + bf16 store (zero the N-pad so it is safe K-pad downstream)
    #pragma unroll
    for (int t = 0; t < TPW; ++t) {
        int tile = tbase + t;
        if (tile >= NTILES) continue;
        int c = tile * 16 + n16;
        float g = (c < NVALID) ? gamma[c] : 0.f;
        float b = (c < NVALID) ? beta[c]  : 0.f;
        #pragma unroll
        for (int mt = 0; mt < MT; ++mt)
            #pragma unroll
            for (int r = 0; r < 8; ++r) {
                int row = mt * 16 + half * 8 + r;
                float y = 0.f;
                if (c < NVALID) {
                    y = (acc[mt][t][r] - smu[row]) * srstd[row] * g + b;
                    y = 0.5f * y * (1.f + erff(y * 0.70710678118f));
                }
                out[(size_t)(row0 + row) * NPAD + c] = f2bf_bits(y);
            }
    }
}

// ---------- final GEMM (g3) + spec bias + noise + top-2 + aux partials ----------
__global__ __launch_bounds__(128)
void out_topk_kernel(const unsigned short* __restrict__ A,      // h2 [B,608] bf16
                     const unsigned short* __restrict__ W,      // g3 [64,608] bf16
                     const float* __restrict__ g3_b,            // [64]
                     const float* __restrict__ spec,            // [64,4]
                     const float* __restrict__ noise,           // [B,64]
                     const float* __restrict__ rp,              // [B,4]
                     const float* __restrict__ nscale,          // [B]
                     float* __restrict__ out_tw,                // [B,2]
                     float* __restrict__ out_idx,               // [B,2] (as float)
                     float* __restrict__ prob_part,             // [gridDim,64]
                     float* __restrict__ load_part)             // [gridDim,64]
{
    __shared__ u32t  ldsA[2][256];
    __shared__ float cleanb[16][64];
    __shared__ float noisyb[16][64];
    __shared__ float rps[16][4];
    __shared__ float nss[16];
    __shared__ float mxs[16], invs[16];
    __shared__ int   idxs[16][2];

    const int tid  = threadIdx.x;
    const int wave = tid >> 5;
    const int lane = tid & 31;
    const int n16  = lane & 15;
    const int half = lane >> 4;
    const int row0 = blockIdx.x * 16;
    const unsigned ldsbase = lds_off32(&ldsA[0][0]);

    if (tid < 64) rps[tid >> 2][tid & 3] = rp[(size_t)(row0 + (tid >> 2)) * 4 + (tid & 3)];
    if (tid < 16) nss[tid] = nscale[row0 + tid];

    const int ar  = tid >> 3;        // row 0..15
    const int ac4 = tid & 7;         // chunk 0..7
    auto issue = [&](int k0, int buf) {
        const void* g = (const void*)(A + (size_t)(row0 + ar) * 608 + k0 + 4 * ac4);
        async_ld_b64(ldsbase + (unsigned)(buf * 1024 + tid * 8), g);
    };

    v8f acc = {0.f,0.f,0.f,0.f,0.f,0.f,0.f,0.f};
    issue(0, 0);
    int cur = 0;
    for (int k0 = 0; k0 < 608; k0 += 32) {
        wait_async0();
        __syncthreads();
        if (k0 + 32 < 608) issue(k0 + 32, cur ^ 1);
        union { v16bf v; u32t u[8]; } af;
        #pragma unroll
        for (int vv = 0; vv < 8; ++vv)
            af.u[vv] = ldsA[cur][n16 * 16 + a_pair_idx(vv, half)];
        v16bf bfv = *(const v16bf*)(W + (size_t)(wave * 16 + n16) * 608 + k0 + half * 16);
        acc = __builtin_amdgcn_wmma_f32_16x16x32_bf16(false, af.v, false, bfv,
                                                      (short)0, acc, false, false);
        cur ^= 1;
    }

    int c = wave * 16 + n16;
    float bia = g3_b[c];
    float s0 = spec[c * 4 + 0], s1 = spec[c * 4 + 1], s2 = spec[c * 4 + 2], s3 = spec[c * 4 + 3];
    #pragma unroll
    for (int r = 0; r < 8; ++r) {
        int row = half * 8 + r;
        float cl = acc[r] + bia + rps[row][0] * s0 + rps[row][1] * s1
                                + rps[row][2] * s2 + rps[row][3] * s3;
        cleanb[row][c] = cl;
        noisyb[row][c] = cl + noise[(size_t)(row0 + row) * 64 + c] * nss[row];
    }
    __syncthreads();

    if (tid < 16) {
        int r = tid;
        float v1 = -3.4e38f, v2 = -3.4e38f; int i1 = 0, i2 = 0;
        for (int e = 0; e < 64; ++e) {
            float v = noisyb[r][e];
            if (v > v1)      { v2 = v1; i2 = i1; v1 = v; i1 = e; }
            else if (v > v2) { v2 = v; i2 = e; }
        }
        float e2 = __expf(v2 - v1);
        float tsum = 1.f + e2;
        size_t ro = (size_t)(row0 + r) * 2;
        out_tw[ro]      = 1.f / tsum;
        out_tw[ro + 1]  = e2 / tsum;
        out_idx[ro]     = (float)i1;
        out_idx[ro + 1] = (float)i2;
        idxs[r][0] = i1; idxs[r][1] = i2;
        float m = -3.4e38f;
        for (int e = 0; e < 64; ++e) m = fmaxf(m, cleanb[r][e]);
        float se = 0.f;
        for (int e = 0; e < 64; ++e) se += __expf(cleanb[r][e] - m);
        mxs[r] = m; invs[r] = 1.f / se;
    }
    __syncthreads();
    if (tid < 64) {
        int e = tid;
        float ps = 0.f, lc = 0.f;
        #pragma unroll 4
        for (int r = 0; r < 16; ++r) {
            ps += __expf(cleanb[r][e] - mxs[r]) * invs[r];
            lc += (idxs[r][0] == e) ? 1.f : 0.f;
            lc += (idxs[r][1] == e) ? 1.f : 0.f;
        }
        prob_part[(size_t)blockIdx.x * 64 + e] = ps;
        load_part[(size_t)blockIdx.x * 64 + e] = lc;
    }
}

// ---------- deterministic aux-loss reduction ----------
__global__ __launch_bounds__(64)
void finalize_kernel(const float* __restrict__ prob_part, const float* __restrict__ load_part,
                     int nblk, float* __restrict__ out_loss)
{
    __shared__ float probsum[64], loadsum[64];
    int e = threadIdx.x;
    float ps = 0.f, ls = 0.f;
    for (int b = 0; b < nblk; ++b) {
        ps += prob_part[(size_t)b * 64 + e];
        ls += load_part[(size_t)b * 64 + e];
    }
    probsum[e] = ps; loadsum[e] = ls;
    __syncthreads();
    if (e == 0) {
        const float invB = 1.f / (float)BROWS;
        float ll = 0.f, sum_imp = 0.f;
        for (int i = 0; i < 64; ++i) {
            ll += (probsum[i] * invB) * (loadsum[i] * invB);
            sum_imp += probsum[i];
        }
        float mean_imp = sum_imp / 64.f;
        float var = 0.f;
        for (int i = 0; i < 64; ++i) { float d = probsum[i] - mean_imp; var += d * d; }
        var /= 63.f;                               // unbiased (torch.var)
        *out_loss = 0.01f * (64.f * ll + var / mean_imp);
    }
}

// ---------- host ----------
extern "C" void kernel_launch(void* const* d_in, const int* in_sizes, int n_in,
                              void* d_out, int out_size, void* d_ws, size_t ws_size,
                              hipStream_t stream)
{
    const float* x      = (const float*)d_in[0];
    const float* noise  = (const float*)d_in[1];
    const float* vol_w  = (const float*)d_in[2];
    const float* vol_b  = (const float*)d_in[3];
    const float* vol_g  = (const float*)d_in[4];
    const float* vol_bb = (const float*)d_in[5];
    const float* trend_w  = (const float*)d_in[6];
    const float* trend_b  = (const float*)d_in[7];
    const float* trend_g  = (const float*)d_in[8];
    const float* trend_bb = (const float*)d_in[9];
    const float* mom_w  = (const float*)d_in[10];
    const float* mom_b  = (const float*)d_in[11];
    const float* mom_g  = (const float*)d_in[12];
    const float* mom_bb = (const float*)d_in[13];
    const float* reg_w  = (const float*)d_in[14];
    const float* reg_b  = (const float*)d_in[15];
    const float* reg_g  = (const float*)d_in[16];
    const float* reg_bb = (const float*)d_in[17];
    const float* rc1_w = (const float*)d_in[18];
    const float* rc1_b = (const float*)d_in[19];
    const float* rc2_w = (const float*)d_in[20];
    const float* rc2_b = (const float*)d_in[21];
    const float* g1_w  = (const float*)d_in[22];
    const float* g1_b  = (const float*)d_in[23];
    const float* g1_g  = (const float*)d_in[24];
    const float* g1_bb = (const float*)d_in[25];
    const float* g2_w  = (const float*)d_in[26];
    const float* g2_b  = (const float*)d_in[27];
    const float* g2_g  = (const float*)d_in[28];
    const float* g2_bb = (const float*)d_in[29];
    const float* g3_w  = (const float*)d_in[30];
    const float* g3_b  = (const float*)d_in[31];
    const float* ns1_w = (const float*)d_in[32];
    const float* ns1_b = (const float*)d_in[33];
    const float* ns2_w = (const float*)d_in[34];
    const float* ns2_b = (const float*)d_in[35];
    const float* spec  = (const float*)d_in[36];

    char* ws = (char*)d_ws;
    auto take = [&](size_t bytes) -> char* {
        char* p = ws;
        ws += (bytes + 255) & ~(size_t)255;
        return p;
    };
    unsigned short* gi    = (unsigned short*)take((size_t)BROWS * 608 * 2);
    unsigned short* h1    = (unsigned short*)take((size_t)BROWS * 1184 * 2);
    unsigned short* h2    = (unsigned short*)take((size_t)BROWS * 608 * 2);
    unsigned short* Wb    = (unsigned short*)take((size_t)64 * 512 * 2);
    unsigned short* g1bf  = (unsigned short*)take((size_t)1184 * 608 * 2);
    unsigned short* g2bf  = (unsigned short*)take((size_t)608 * 1184 * 2);
    unsigned short* g3bf  = (unsigned short*)take((size_t)64 * 608 * 2);
    float* rp_ws     = (float*)take((size_t)BROWS * 4 * 4);
    float* nscale_ws = (float*)take((size_t)BROWS * 4);
    float* enc_b     = (float*)take(64 * 4);
    float* enc_g     = (float*)take(64 * 4);
    float* enc_bb    = (float*)take(64 * 4);
    const int NBLK16 = BROWS / 16;   // 1024
    const int NBLK32 = BROWS / 32;   // 512
    float* prob_part = (float*)take((size_t)NBLK16 * 64 * 4);
    float* load_part = (float*)take((size_t)NBLK16 * 64 * 4);

    float* out_tw   = (float*)d_out;
    float* out_idx  = (float*)d_out + (size_t)BROWS * 2;
    float* out_loss = (float*)d_out + (size_t)BROWS * 4;

    // weight prep
    cvt_pad_kernel<<<32, 256, 0, stream>>>(vol_w,   Wb + 0 * 16 * 512, 16, 512, 16, 512);
    cvt_pad_kernel<<<32, 256, 0, stream>>>(trend_w, Wb + 1 * 16 * 512, 16, 512, 16, 512);
    cvt_pad_kernel<<<32, 256, 0, stream>>>(mom_w,   Wb + 2 * 16 * 512, 16, 512, 16, 512);
    cvt_pad_kernel<<<32, 256, 0, stream>>>(reg_w,   Wb + 3 * 16 * 512, 16, 512, 16, 512);
    cvt_pad_kernel<<<1408, 256, 0, stream>>>(g1_w, g1bf, 1160, 580, 1184, 608);
    cvt_pad_kernel<<<1408, 256, 0, stream>>>(g2_w, g2bf, 580, 1160, 608, 1184);
    cvt_pad_kernel<<<152, 256, 0, stream>>>(g3_w, g3bf, 64, 580, 64, 608);
    init_kernel<<<1, 64, 0, stream>>>(vol_b, vol_g, vol_bb, trend_b, trend_g, trend_bb,
                                      mom_b, mom_g, mom_bb, reg_b, reg_g, reg_bb,
                                      enc_b, enc_g, enc_bb);

    // pipeline
    encoder_kernel<<<NBLK16, 128, 0, stream>>>(x, Wb, enc_b, enc_g, enc_bb,
                                               rc1_w, rc1_b, rc2_w, rc2_b,
                                               ns1_w, ns1_b, ns2_w, ns2_b,
                                               gi, rp_ws, nscale_ws);
    gemm_ln_gelu_kernel<1184, 608, 1160, 8, 10, 2><<<NBLK32, 256, 0, stream>>>(
        gi, g1bf, g1_b, g1_g, g1_bb, h1);
    gemm_ln_gelu_kernel<608, 1184, 580, 8, 5, 2><<<NBLK32, 256, 0, stream>>>(
        h1, g2bf, g2_b, g2_g, g2_bb, h2);
    out_topk_kernel<<<NBLK16, 128, 0, stream>>>(h2, g3bf, g3_b, spec, noise,
                                                rp_ws, nscale_ws,
                                                out_tw, out_idx, prob_part, load_part);
    finalize_kernel<<<1, 64, 0, stream>>>(prob_part, load_part, NBLK16, out_loss);
}
